// MambaBlock_80771154968656
// MI455X (gfx1250) — matrix operand order repaired and checked by reading, not compiled, not run on hardware
//
#include <hip/hip_runtime.h>
#include <hip/hip_bf16.h>
#include <math.h>

// ---------------------------------------------------------------------------
// Mamba block + MLP for MI455X (gfx1250).
// bf16 WMMA GEMMs (fp32 accum), register-blocked 32x64 per wave,
// weight tiles staged into LDS with async global->LDS copies (double buffer).
// ---------------------------------------------------------------------------

typedef __attribute__((ext_vector_type(16))) __bf16 v16bf;
typedef __attribute__((ext_vector_type(8)))  __bf16 v8bf;
typedef __attribute__((ext_vector_type(8)))  float  v8f;

#define BSZ     2
#define SEQ     2048
#define DIM     1024
#define MLPD    4096
#define HALF    1024
#define DINNER  2048
#define DSTATE  16
#define DTRANK  64
#define NTOK    (BSZ * SEQ)      // 4096 tokens

// Epilogue flag bits
#define EP_BIAS     1
#define EP_GELU     2
#define EP_SOFTPLUS 4
#define EP_RES      8
#define EP_OUTF     16
#define EP_OUTB     32

// ---------------------------------------------------------------------------
// A-fragment loader: lane m = l&15; K-half = lane>>4;
// per-lane K = 16*half + {0..15}  ->  two 16-byte loads + shuffle.
// ---------------------------------------------------------------------------
__device__ __forceinline__ v16bf load_a_frag(const __bf16* p, int half) {
    v8bf lo = *(const v8bf*)(p + half * 16);
    v8bf hi = *(const v8bf*)(p + 8 + half * 16);
    return __builtin_shufflevector(lo, hi,
               0, 1, 2, 3, 4, 5, 6, 7, 8, 9, 10, 11, 12, 13, 14, 15);
}

__device__ __forceinline__ float epilogue_apply(int EPI, float v) {
    if (EPI & EP_GELU)     v = 0.5f * v * (1.0f + erff(v * 0.70710678118654752f));
    if (EPI & EP_SOFTPLUS) v = (v > 20.0f) ? v : log1pf(__expf(v));
    return v;
}

// ---------------------------------------------------------------------------
// Blocked GEMM: C[M,N] = A[M,K] * W[N,K]^T, fp32 accumulate.
//  - block = 256 threads (8 waves) -> 256-row x 64-col macro tile
//  - wave  = 32x64 output (2 M-tiles x 4 N-tiles), K step 32
//  - B tile (64x32 bf16 = 4KB) shared by all 8 waves, staged in LDS via
//    global_load_async_to_lds_b128, double buffered (ASYNCcnt pipeline)
// grid: (N/64, M/256).  Requires N%64==0, K%32==0, M%256==0.
// ---------------------------------------------------------------------------
template <int EPI>
__global__ void gemm_wmma_bf16_blk(const __bf16* __restrict__ A, int lda,
                                   const __bf16* __restrict__ W,
                                   const float* __restrict__ bias,
                                   const float* __restrict__ res,
                                   float* __restrict__ Cf,
                                   __bf16* __restrict__ Cb,
                                   int N, int K) {
    __shared__ __align__(128) unsigned char sB[2 * 4096];   // 2 x (64 rows * 64B)

    const int tid  = threadIdx.x;
    const int lane = tid & 31;
    const int wave = tid >> 5;
    const int l15  = lane & 15;
    const int half = lane >> 4;
    const int nb   = blockIdx.x;   // 64-col group
    const int rb   = blockIdx.y;   // 256-row group

    // --- async copy mapping: thread -> 16B chunk of the 64x32 B tile -------
    const int crow   = tid >> 2;             // 0..63  (row of W within tile)
    const int cchunk = tid & 3;              // 0..3   (16B chunk within 64B row)
    const __bf16*  gsrc_row = W + (size_t)(nb * 64 + crow) * K;
    const uint32_t lds_base = (uint32_t)(uintptr_t)sB;

    auto issue_copy = [&](int buf, int k0) {
        uint32_t dst = lds_base + (uint32_t)(buf * 4096 + crow * 64 + cchunk * 16);
        uint64_t src = (uint64_t)(uintptr_t)gsrc_row + (uint64_t)k0 * 2 + cchunk * 16;
        asm volatile("global_load_async_to_lds_b128 %0, %1, off"
                     :: "v"(dst), "v"(src) : "memory");
    };

    // --- per-wave A row pointers ------------------------------------------
    const __bf16* arow0 = A + (size_t)(rb * 256 + wave * 32 + l15) * lda;
    const __bf16* arow1 = arow0 + (size_t)16 * lda;

    v8f acc[2][4];
#pragma unroll
    for (int mi = 0; mi < 2; ++mi)
#pragma unroll
        for (int ni = 0; ni < 4; ++ni) acc[mi][ni] = (v8f){};

    const int steps = K >> 5;
    issue_copy(0, 0);

    for (int s = 0; s < steps; ++s) {
        const int k0 = s << 5;
        if (s + 1 < steps) {
            issue_copy((s + 1) & 1, k0 + 32);
            asm volatile("s_wait_asynccnt 1" ::: "memory");  // oldest copy done
        } else {
            asm volatile("s_wait_asynccnt 0" ::: "memory");
        }
        __syncthreads();                                     // tile visible to all

        v16bf a0 = load_a_frag(arow0 + k0, half);
        v16bf a1 = load_a_frag(arow1 + k0, half);
        const unsigned char* bufp = sB + ((s & 1) * 4096);
#pragma unroll
        for (int ni = 0; ni < 4; ++ni) {
            v16bf b = *(const v16bf*)(bufp + (ni * 16 + l15) * 64 + half * 32);
            acc[0][ni] = __builtin_amdgcn_wmma_f32_16x16x32_bf16(
                             false, a0, false, b, (short)0, acc[0][ni], false, false);
            acc[1][ni] = __builtin_amdgcn_wmma_f32_16x16x32_bf16(
                             false, a1, false, b, (short)0, acc[1][ni], false, false);
        }
        __syncthreads();                                     // done reading buffer
    }

    // --- epilogue ----------------------------------------------------------
#pragma unroll
    for (int mi = 0; mi < 2; ++mi) {
        const int mbase = rb * 256 + wave * 32 + mi * 16 + half * 8;
#pragma unroll
        for (int ni = 0; ni < 4; ++ni) {
            const int col = nb * 64 + ni * 16 + l15;
            float bv = 0.0f;
            if (EPI & EP_BIAS) bv = bias[col];
#pragma unroll
            for (int r = 0; r < 8; ++r) {
                size_t idx = (size_t)(mbase + r) * N + col;
                float  v   = epilogue_apply(EPI, acc[mi][ni][r] + bv);
                if (EPI & EP_RES)  v += res[idx];
                if (EPI & EP_OUTF) Cf[idx] = v;
                if (EPI & EP_OUTB) Cb[idx] = (__bf16)v;
            }
        }
    }
}

// ---------------------------------------------------------------------------
// Simple GEMM (one 16x16 tile per wave) for shapes with N%64 != 0 (x_proj).
// ---------------------------------------------------------------------------
template <int EPI>
__global__ void gemm_wmma_bf16(const __bf16* __restrict__ A, int lda,
                               const __bf16* __restrict__ W,
                               const float* __restrict__ bias,
                               const float* __restrict__ res,
                               float* __restrict__ Cf,
                               __bf16* __restrict__ Cb,
                               int M, int N, int K) {
    const int lane   = threadIdx.x & 31;
    const int wave   = threadIdx.x >> 5;
    const int tilesN = N >> 4;
    const int tile   = blockIdx.x * (blockDim.x >> 5) + wave;
    const int total  = (M >> 4) * tilesN;
    if (tile >= total) return;                // wave-uniform: EXEC stays all-1

    const int tm   = tile / tilesN;
    const int tn   = tile % tilesN;
    const int half = lane >> 4;
    const int l15  = lane & 15;

    const __bf16* arow = A + (size_t)(tm * 16 + l15) * lda;
    const __bf16* brow = W + (size_t)(tn * 16 + l15) * K;

    v8f c = {};
    for (int k0 = 0; k0 < K; k0 += 32) {
        __builtin_prefetch(brow + k0 + 512, 0, 1);
        v16bf a = load_a_frag(arow + k0, half);
        v16bf b = *(const v16bf*)(brow + k0 + half * 16);
        c = __builtin_amdgcn_wmma_f32_16x16x32_bf16(
                false, a, false, b, (short)0, c, false, false);
    }

    const int ncol = tn * 16 + l15;
    float bv = 0.0f;
    if (EPI & EP_BIAS) bv = bias[ncol];
#pragma unroll
    for (int r = 0; r < 8; ++r) {
        int    m   = tm * 16 + half * 8 + r;
        size_t idx = (size_t)m * N + ncol;
        float  v   = epilogue_apply(EPI, c[r] + bv);
        if (EPI & EP_RES)  v += res[idx];
        if (EPI & EP_OUTF) Cf[idx] = v;
        if (EPI & EP_OUTB) Cb[idx] = (__bf16)v;
    }
}

// ---------------------------------------------------------------------------
// LayerNorm over last dim (N=1024); one block (256 threads) per row.
// ---------------------------------------------------------------------------
__global__ void layernorm_k(const float* __restrict__ x,
                            const float* __restrict__ g,
                            const float* __restrict__ b,
                            float* __restrict__ of,
                            __bf16* __restrict__ ob, int N) {
    const int    row = blockIdx.x;
    const float* xr  = x + (size_t)row * N;
    float s = 0.0f, s2 = 0.0f;
    for (int i = threadIdx.x; i < N; i += blockDim.x) {
        float v = xr[i];
        s += v;
        s2 += v * v;
    }
    __shared__ float rs[256], rq[256];
    rs[threadIdx.x] = s;
    rq[threadIdx.x] = s2;
    __syncthreads();
    for (int o = blockDim.x >> 1; o > 0; o >>= 1) {
        if ((int)threadIdx.x < o) {
            rs[threadIdx.x] += rs[threadIdx.x + o];
            rq[threadIdx.x] += rq[threadIdx.x + o];
        }
        __syncthreads();
    }
    const float mu  = rs[0] / (float)N;
    const float var = rq[0] / (float)N - mu * mu;
    const float inv = rsqrtf(var + 1e-5f);
    for (int i = threadIdx.x; i < N; i += blockDim.x) {
        float v = (xr[i] - mu) * inv * g[i] + b[i];
        if (of) of[(size_t)row * N + i] = v;
        if (ob) ob[(size_t)row * N + i] = (__bf16)v;
    }
}

// ---------------------------------------------------------------------------
// float -> bf16 conversion, grid-stride.
// ---------------------------------------------------------------------------
__global__ void f2bf_k(const float* __restrict__ s, __bf16* __restrict__ d,
                       size_t n) {
    size_t i  = (size_t)blockIdx.x * blockDim.x + threadIdx.x;
    size_t st = (size_t)gridDim.x * blockDim.x;
    for (; i < n; i += st) d[i] = (__bf16)s[i];
}

// ---------------------------------------------------------------------------
// Depthwise conv (K=4, pad left 1 / right 2) along L + SiLU.
// ---------------------------------------------------------------------------
__global__ void dwconv_silu_k(const float* __restrict__ xz, int chan_off,
                              const float* __restrict__ w,     // [1024,4]
                              const float* __restrict__ bias,  // [1024]
                              float* __restrict__ of,
                              __bf16* __restrict__ ob, int ob_ld, int ob_off) {
    int idx = blockIdx.x * blockDim.x + threadIdx.x;  // over NTOK*HALF
    int d   = idx & (HALF - 1);
    int t   = idx >> 10;            // token = b*SEQ + l
    int bb  = t >> 11;              // SEQ = 2048
    int l   = t & (SEQ - 1);
    float acc = bias[d];
#pragma unroll
    for (int k = 0; k < 4; ++k) {
        int ll = l + k - 1;
        if (ll >= 0 && ll < SEQ)
            acc += xz[(size_t)(bb * SEQ + ll) * DINNER + chan_off + d] * w[d * 4 + k];
    }
    float sv = acc / (1.0f + __expf(-acc));  // SiLU
    if (of) of[(size_t)t * HALF + d] = sv;
    if (ob) ob[(size_t)t * ob_ld + ob_off + d] = (__bf16)sv;
}

// ---------------------------------------------------------------------------
// Selective scan: one thread per (batch, channel); N=16 state in registers.
// ---------------------------------------------------------------------------
__global__ void scan_k(const float* __restrict__ delta,  // [token,1024]
                       const float* __restrict__ u,      // [token,1024]
                       const float* __restrict__ xdbl,   // [token,96]
                       const float* __restrict__ A_log,  // [1024,16]
                       const float* __restrict__ Dp,     // [1024]
                       __bf16* __restrict__ y2) {        // [token,2048]
    int gid = blockIdx.x * blockDim.x + threadIdx.x;     // 0..2047
    int ch  = gid & (HALF - 1);
    int bb  = gid >> 10;
    float A[DSTATE], h[DSTATE];
#pragma unroll
    for (int n = 0; n < DSTATE; ++n) {
        A[n] = -__expf(A_log[ch * DSTATE + n]);
        h[n] = 0.0f;
    }
    const float D = Dp[ch];
    for (int l = 0; l < SEQ; ++l) {
        size_t t  = (size_t)bb * SEQ + l;
        float  dl = delta[t * HALF + ch];
        float  uv = u[t * HALF + ch];
        const float* Bv = xdbl + t * 96 + DTRANK;
        const float* Cv = Bv + DSTATE;
        float du  = dl * uv;
        float acc = 0.0f;
#pragma unroll
        for (int n = 0; n < DSTATE; ++n) {
            h[n] = __expf(dl * A[n]) * h[n] + du * Bv[n];
            acc += h[n] * Cv[n];
        }
        y2[t * DINNER + ch] = (__bf16)(acc + uv * D);
    }
}

// ---------------------------------------------------------------------------
// Host-side launch
// ---------------------------------------------------------------------------
extern "C" void kernel_launch(void* const* d_in, const int* in_sizes, int n_in,
                              void* d_out, int out_size, void* d_ws, size_t ws_size,
                              hipStream_t stream) {
    const float* x        = (const float*)d_in[0];
    const float* ln1_g    = (const float*)d_in[1];
    const float* ln1_b    = (const float*)d_in[2];
    const float* in_proj  = (const float*)d_in[3];   // [2048,1024]
    const float* conv_x_w = (const float*)d_in[4];   // [1024,1,4]
    const float* conv_x_b = (const float*)d_in[5];
    const float* conv_z_w = (const float*)d_in[6];
    const float* conv_z_b = (const float*)d_in[7];
    const float* x_proj   = (const float*)d_in[8];   // [96,1024]
    const float* dt_proj  = (const float*)d_in[9];   // [1024,64]
    const float* dt_b     = (const float*)d_in[10];
    const float* A_log    = (const float*)d_in[11];  // [1024,16]
    const float* Dp       = (const float*)d_in[12];
    const float* out_proj = (const float*)d_in[13];  // [1024,2048]
    const float* ln2_g    = (const float*)d_in[14];
    const float* ln2_b    = (const float*)d_in[15];
    const float* mlp_w1   = (const float*)d_in[16];  // [4096,1024]
    const float* mlp_b1   = (const float*)d_in[17];
    const float* mlp_w2   = (const float*)d_in[18];  // [1024,4096]
    const float* mlp_b2   = (const float*)d_in[19];
    float*       out      = (float*)d_out;

    // --- workspace carve ---------------------------------------------------
    size_t off = 0;
    auto alloc = [&](size_t bytes) -> void* {
        void* p = (char*)d_ws + off;
        off += (bytes + 255) & ~(size_t)255;
        return p;
    };
    __bf16* bW_in  = (__bf16*)alloc((size_t)DINNER * DIM * 2);
    __bf16* bW_xp  = (__bf16*)alloc((size_t)96 * DIM * 2);
    __bf16* bW_dt  = (__bf16*)alloc((size_t)HALF * DTRANK * 2);
    __bf16* bW_out = (__bf16*)alloc((size_t)DIM * DINNER * 2);
    __bf16* bW_m1  = (__bf16*)alloc((size_t)MLPD * DIM * 2);
    __bf16* bW_m2  = (__bf16*)alloc((size_t)DIM * MLPD * 2);
    __bf16* b_xn   = (__bf16*)alloc((size_t)NTOK * DIM * 2);
    float*  f_xz   = (float*) alloc((size_t)NTOK * DINNER * 4);
    float*  f_u    = (float*) alloc((size_t)NTOK * HALF * 4);
    __bf16* b_u    = (__bf16*)alloc((size_t)NTOK * HALF * 2);
    float*  f_xdbl = (float*) alloc((size_t)NTOK * 96 * 4);
    __bf16* b_xdbl = (__bf16*)alloc((size_t)NTOK * 96 * 2);
    float*  f_dlt  = (float*) alloc((size_t)NTOK * HALF * 4);
    __bf16* b_y2   = (__bf16*)alloc((size_t)NTOK * DINNER * 2);
    float*  f_x1   = (float*) alloc((size_t)NTOK * DIM * 4);
    __bf16* b_h    = (__bf16*)alloc((size_t)NTOK * DIM * 2);
    __bf16* b_h2   = (__bf16*)alloc((size_t)NTOK * MLPD * 2);
    (void)ws_size; (void)in_sizes; (void)n_in; (void)out_size;

    const int T = 256;

    // --- weights -> bf16 ---------------------------------------------------
    f2bf_k<<<1024, T, 0, stream>>>(in_proj,  bW_in,  (size_t)DINNER * DIM);
    f2bf_k<<<128,  T, 0, stream>>>(x_proj,   bW_xp,  (size_t)96 * DIM);
    f2bf_k<<<64,   T, 0, stream>>>(dt_proj,  bW_dt,  (size_t)HALF * DTRANK);
    f2bf_k<<<1024, T, 0, stream>>>(out_proj, bW_out, (size_t)DIM * DINNER);
    f2bf_k<<<2048, T, 0, stream>>>(mlp_w1,   bW_m1,  (size_t)MLPD * DIM);
    f2bf_k<<<2048, T, 0, stream>>>(mlp_w2,   bW_m2,  (size_t)DIM * MLPD);

    // --- LN1 -> bf16 -------------------------------------------------------
    layernorm_k<<<NTOK, T, 0, stream>>>(x, ln1_g, ln1_b, nullptr, b_xn, DIM);

    // --- in_proj: xz = xn @ W_in^T   [4096,2048] ---------------------------
    gemm_wmma_bf16_blk<EP_OUTF>
        <<<dim3(DINNER / 64, NTOK / 256), T, 0, stream>>>(
        b_xn, DIM, bW_in, nullptr, nullptr, f_xz, nullptr, DINNER, DIM);

    // --- depthwise conv + SiLU (x path -> u f32+bf16, z path -> y2[:,1024:])
    dwconv_silu_k<<<(NTOK * HALF) / T, T, 0, stream>>>(
        f_xz, 0, conv_x_w, conv_x_b, f_u, b_u, HALF, 0);
    dwconv_silu_k<<<(NTOK * HALF) / T, T, 0, stream>>>(
        f_xz, HALF, conv_z_w, conv_z_b, nullptr, b_y2, DINNER, HALF);

    // --- x_proj: x_dbl = u @ W_xp^T   [4096,96] (f32 + bf16), N%64!=0 ------
    {
        int tiles  = (NTOK / 16) * (96 / 16);
        int blocks = (tiles + 7) / 8;
        gemm_wmma_bf16<EP_OUTF | EP_OUTB><<<blocks, T, 0, stream>>>(
            b_u, HALF, bW_xp, nullptr, nullptr, f_xdbl, b_xdbl, NTOK, 96, DIM);
    }

    // --- dt_proj + softplus: delta = softplus(dt @ W_dt^T + b) [4096,1024] -
    gemm_wmma_bf16_blk<EP_BIAS | EP_SOFTPLUS | EP_OUTF>
        <<<dim3(HALF / 64, NTOK / 256), T, 0, stream>>>(
        b_xdbl, 96, bW_dt, dt_b, nullptr, f_dlt, nullptr, HALF, DTRANK);

    // --- selective scan -> y2[:, :1024] (bf16) -----------------------------
    scan_k<<<(BSZ * HALF) / T, T, 0, stream>>>(f_dlt, f_u, f_xdbl, A_log, Dp, b_y2);

    // --- out_proj + residual: x1 = x + y2 @ W_out^T  [4096,1024] -----------
    gemm_wmma_bf16_blk<EP_RES | EP_OUTF>
        <<<dim3(DIM / 64, NTOK / 256), T, 0, stream>>>(
        b_y2, DINNER, bW_out, nullptr, x, f_x1, nullptr, DIM, DINNER);

    // --- LN2 -> bf16 -------------------------------------------------------
    layernorm_k<<<NTOK, T, 0, stream>>>(f_x1, ln2_g, ln2_b, nullptr, b_h, DIM);

    // --- MLP up + exact GELU -> bf16 [4096,4096] ---------------------------
    gemm_wmma_bf16_blk<EP_BIAS | EP_GELU | EP_OUTB>
        <<<dim3(MLPD / 64, NTOK / 256), T, 0, stream>>>(
        b_h, DIM, bW_m1, mlp_b1, nullptr, nullptr, b_h2, MLPD, DIM);

    // --- MLP down + bias + residual -> out [4096,1024] ---------------------
    gemm_wmma_bf16_blk<EP_BIAS | EP_RES | EP_OUTF>
        <<<dim3(DIM / 64, NTOK / 256), T, 0, stream>>>(
        b_h2, MLPD, bW_m2, mlp_b2, f_x1, out, nullptr, DIM, MLPD);
}
